// MultiHeadAttention_60301340836272
// MI455X (gfx1250) — matrix operand order repaired and checked by reading, not compiled, not run on hardware
//
#include <hip/hip_runtime.h>
#include <hip/hip_bf16.h>

#define D_MODEL 1024
#define S_LEN   2048
#define BATCH   2
#define NHEAD   16
#define DK      64

typedef __attribute__((ext_vector_type(16))) __bf16 bf16x16;
typedef __attribute__((ext_vector_type(8)))  float  f32x8;
typedef __attribute__((ext_vector_type(4)))  unsigned int u32x4;
typedef __attribute__((ext_vector_type(4)))  int i32x4;
typedef __attribute__((ext_vector_type(8)))  int i32x8;

#if defined(__has_builtin)
#if __has_builtin(__builtin_amdgcn_tensor_load_to_lds)
#define USE_TDM 1
#endif
#endif

union FragAB {
    bf16x16 v;
    uint4   q[2];
};

__device__ __forceinline__ unsigned short f32_to_bf16(float f) {
    union { float f; unsigned int u; } a;
    a.f = f;
    unsigned int u = a.u;
    u += 0x7FFFu + ((u >> 16) & 1u);   // round-to-nearest-even
    return (unsigned short)(u >> 16);
}

__device__ __forceinline__ f32x8 zero8() {
    f32x8 z;
#pragma unroll
    for (int i = 0; i < 8; ++i) z[i] = 0.0f;
    return z;
}

#ifdef USE_TDM
// ---------------------------------------------------------------------------
// TDM: DMA a [128 rows x 32 bf16] tile (row stride 1024 bf16) from global to
// LDS, hardware-padding each 64B row with +16B so LDS row stride = 40 bf16.
// D# group0/group1 packed per CDNA5 ISA 8.3/8.4.
// ---------------------------------------------------------------------------
__device__ __forceinline__ void tdm_load_tile(const unsigned short* gtile,
                                              unsigned int lds_byte,
                                              unsigned int tensor_rows) {
    unsigned long long ga = (unsigned long long)(uintptr_t)gtile;
    u32x4 g0;
    g0[0] = 1u;                                        // count=1, user mode
    g0[1] = lds_byte;                                  // lds_addr
    g0[2] = (unsigned int)(ga & 0xFFFFFFFFu);          // global_addr[31:0]
    g0[3] = (unsigned int)((ga >> 32) & 0x01FFFFFFu)   // global_addr[56:32]
          | (2u << 30);                                // type = 2 ("image")
    i32x8 g1;
    // workgroup_mask=0 | data_size=1(2B)<<16 | pad_enable<<20
    // | pad_interval=3 (16 DWORD = 64B)<<22 | pad_amount=3 (4 DWORD = 16B)<<25
    g1[0] = (int)(0x00010000u | (1u << 20) | (3u << 22) | (3u << 25));
    g1[1] = (int)(1024u << 16);                        // tensor_dim0[15:0]
    g1[2] = (int)((1024u >> 16) | (tensor_rows << 16));// dim0[31:16]|dim1[15:0]
    g1[3] = (int)((tensor_rows >> 16) | (32u << 16));  // dim1[31:16]|tile_dim0
    g1[4] = 128;                                       // tile_dim1 (tile_dim2=0)
    g1[5] = 1024;                                      // tensor_dim0_stride lo
    g1[6] = 0;                                         // stride0 hi | stride1 lo
    g1[7] = 0;
    i32x4 z4;
#pragma unroll
    for (int i = 0; i < 4; ++i) z4[i] = 0;
#if defined(__clang_major__) && (__clang_major__ >= 23)
    i32x8 z8;
#pragma unroll
    for (int i = 0; i < 8; ++i) z8[i] = 0;
    __builtin_amdgcn_tensor_load_to_lds(g0, g1, z4, z4, z8, 0);
#else
    __builtin_amdgcn_tensor_load_to_lds(g0, g1, z4, z4, 0);
#endif
}
#endif // USE_TDM

// ---------------------------------------------------------------------------
// fp32 -> bf16 conversion (8 elements / thread, 16B-aligned vector traffic)
// ---------------------------------------------------------------------------
__global__ __launch_bounds__(256)
void cvt_f32_bf16_kernel(const float* __restrict__ in,
                         unsigned short* __restrict__ out, int n) {
    int i = (blockIdx.x * 256 + threadIdx.x) * 8;
    if (i >= n) return;
    float4 f0 = *reinterpret_cast<const float4*>(in + i);
    float4 f1 = *reinterpret_cast<const float4*>(in + i + 4);
    unsigned short r[8];
    r[0] = f32_to_bf16(f0.x); r[1] = f32_to_bf16(f0.y);
    r[2] = f32_to_bf16(f0.z); r[3] = f32_to_bf16(f0.w);
    r[4] = f32_to_bf16(f1.x); r[5] = f32_to_bf16(f1.y);
    r[6] = f32_to_bf16(f1.z); r[7] = f32_to_bf16(f1.w);
    uint4 o;
    o.x = (unsigned)r[0] | ((unsigned)r[1] << 16);
    o.y = (unsigned)r[2] | ((unsigned)r[3] << 16);
    o.z = (unsigned)r[4] | ((unsigned)r[5] << 16);
    o.w = (unsigned)r[6] | ((unsigned)r[7] << 16);
    *reinterpret_cast<uint4*>(out + i) = o;
}

// ---------------------------------------------------------------------------
// Y[M,N] = X[M,K]*W[N,K]^T + bias.  X,W bf16, K=N=1024, M=4096.
// Block = 256 thr (8 waves), tile 128M x 128N, K staged 32-wide via LDS
// (TDM DMA when available, manual b128 copies otherwise).
// MODE 0: fp32 flat [m*1024+n]           (output projection -> d_out)
// MODE 1: bf16 head-major [b,h,s,d]      (Q, K)
// MODE 2: bf16 head-transposed [b,h,d,s] (V)
// ---------------------------------------------------------------------------
template <int MODE>
__global__ __launch_bounds__(256)
void gemm_bias_kernel(const unsigned short* __restrict__ X,
                      const unsigned short* __restrict__ W,
                      const float* __restrict__ bias,
                      void* __restrict__ Yout) {
    constexpr int K = D_MODEL;
    constexpr int LR = 40;                    // 32 + 8 pad bf16 per LDS row
    __shared__ unsigned short As[128 * LR];
    __shared__ unsigned short Bs[128 * LR];

    const int m0    = blockIdx.y * 128;
    const int n0    = blockIdx.x * 128;
    const int tid   = threadIdx.x;
    const int wave  = tid >> 5;
    const int lane  = tid & 31;
    const int l15   = lane & 15;
    const int lhalf = lane >> 4;

#ifndef USE_TDM
    // LDS fill assignment: thread -> row (0..127), 16B chunk pair
    const int frow = tid >> 1;
    const int fcol = (tid & 1) * 8;           // bf16 element offset 0 / 8
#endif

    f32x8 acc[8];
#pragma unroll
    for (int t = 0; t < 8; ++t) acc[t] = zero8();

    for (int kc = 0; kc < K; kc += 32) {
        __syncthreads();                      // previous iter's LDS reads done
#ifdef USE_TDM
        if (wave == 0) {
            tdm_load_tile(X + (size_t)m0 * K + kc,
                          (unsigned int)(uintptr_t)(void*)As, 4096u);
            tdm_load_tile(W + (size_t)n0 * K + kc,
                          (unsigned int)(uintptr_t)(void*)Bs, 1024u);
            __builtin_amdgcn_s_wait_tensorcnt(0);
        }
#else
        {
            const uint4* gx =
                reinterpret_cast<const uint4*>(X + (size_t)(m0 + frow) * K + kc + fcol);
            const uint4* gw =
                reinterpret_cast<const uint4*>(W + (size_t)(n0 + frow) * K + kc + fcol);
            uint4 a0 = gx[0], a1 = gx[2];     // +0B and +32B (16 elements)
            uint4 b0 = gw[0], b1 = gw[2];
            *reinterpret_cast<uint4*>(&As[frow * LR + fcol])      = a0;
            *reinterpret_cast<uint4*>(&As[frow * LR + fcol + 16]) = a1;
            *reinterpret_cast<uint4*>(&Bs[frow * LR + fcol])      = b0;
            *reinterpret_cast<uint4*>(&Bs[frow * LR + fcol + 16]) = b1;
        }
#endif
        __syncthreads();

        // A fragment: 16x32, lane holds row l15, K half-split by 8
        FragAB a;
        {
            const unsigned short* p = &As[(wave * 16 + l15) * LR + lhalf * 8];
            a.q[0] = *reinterpret_cast<const uint4*>(p);
            a.q[1] = *reinterpret_cast<const uint4*>(p + 16);
        }
#pragma unroll
        for (int t = 0; t < 8; ++t) {
            // B fragment: 32x16, lane holds col (W-row) l15, K half-split by 8
            FragAB b;
            const unsigned short* p = &Bs[(t * 16 + l15) * LR + lhalf * 8];
            b.q[0] = *reinterpret_cast<const uint4*>(p);
            b.q[1] = *reinterpret_cast<const uint4*>(p + 16);
            acc[t] = __builtin_amdgcn_wmma_f32_16x16x32_bf16(
                false, a.v, false, b.v, (short)0, acc[t], false, false);
        }
    }

    // Epilogue: C/D layout row = e + 8*lhalf, col = l15
#pragma unroll
    for (int t = 0; t < 8; ++t) {
        const int n  = n0 + t * 16 + l15;
        const float bn = bias[n];
#pragma unroll
        for (int e = 0; e < 8; ++e) {
            const int m = m0 + wave * 16 + 8 * lhalf + e;
            const float y = acc[t][e] + bn;
            if (MODE == 0) {
                reinterpret_cast<float*>(Yout)[(size_t)m * D_MODEL + n] = y;
            } else {
                const int b = m >> 11, s = m & (S_LEN - 1);
                const int h = n >> 6,  d = n & (DK - 1);
                if (MODE == 1) {
                    reinterpret_cast<unsigned short*>(Yout)
                        [((size_t)(b * NHEAD + h) * S_LEN + s) * DK + d] = f32_to_bf16(y);
                } else {
                    reinterpret_cast<unsigned short*>(Yout)
                        [((size_t)(b * NHEAD + h) * DK + d) * S_LEN + s] = f32_to_bf16(y);
                }
            }
        }
    }
}

// ---------------------------------------------------------------------------
// Flash attention, causal. Block = 128 thr (4 waves), 64 query rows / block,
// one 16-row query tile per wave; streams 32-key chunks. Qh/Kh: [B*H,S,64]
// bf16, Vt: [B*H,64,S] bf16, Oh: [B,S,1024] bf16 (head-concat layout).
// ---------------------------------------------------------------------------
__global__ __launch_bounds__(128)
void flash_attn_kernel(const unsigned short* __restrict__ Qh,
                       const unsigned short* __restrict__ Kh,
                       const unsigned short* __restrict__ Vt,
                       unsigned short* __restrict__ Oh) {
    __shared__ unsigned short Pbuf[4][16 * 40];   // per-wave P re-layout tile

    const int bh    = blockIdx.x;                 // 0..31
    const int tid   = threadIdx.x;
    const int wave  = tid >> 5;
    const int lane  = tid & 31;
    const int l15   = lane & 15;
    const int lhalf = lane >> 4;
    const int q0    = blockIdx.y * 64 + wave * 16;

    const unsigned short* Qb = Qh + (size_t)bh * S_LEN * DK;
    const unsigned short* Kb = Kh + (size_t)bh * S_LEN * DK;
    const unsigned short* Vb = Vt + (size_t)bh * DK * S_LEN;

    // Q fragments for d-chunks 0..31 and 32..63 (A layout)
    FragAB qf[2];
#pragma unroll
    for (int c = 0; c < 2; ++c) {
        const unsigned short* p = Qb + (size_t)(q0 + l15) * DK + c * 32 + lhalf * 8;
        qf[c].q[0] = *reinterpret_cast<const uint4*>(p);
        qf[c].q[1] = *reinterpret_cast<const uint4*>(p + 16);
    }

    f32x8 o[4];
#pragma unroll
    for (int t = 0; t < 4; ++t) o[t] = zero8();
    float row_m[8], row_l[8];
#pragma unroll
    for (int e = 0; e < 8; ++e) { row_m[e] = -1.0e30f; row_l[e] = 0.0f; }

    const float scale = 0.125f;                   // 1/sqrt(64)
    const int   kend  = q0 + 16;                  // causal bound for this tile

    for (int kj = 0; kj < kend; kj += 32) {
        // ---- scores: S = Q * K^T for 32 keys (two 16-key N tiles) ----
        f32x8 s0 = zero8(), s1 = zero8();
#pragma unroll
        for (int c = 0; c < 2; ++c) {
            FragAB kf0, kf1;
            const unsigned short* p0 = Kb + (size_t)(kj + l15) * DK      + c * 32 + lhalf * 8;
            const unsigned short* p1 = Kb + (size_t)(kj + 16 + l15) * DK + c * 32 + lhalf * 8;
            kf0.q[0] = *reinterpret_cast<const uint4*>(p0);
            kf0.q[1] = *reinterpret_cast<const uint4*>(p0 + 16);
            kf1.q[0] = *reinterpret_cast<const uint4*>(p1);
            kf1.q[1] = *reinterpret_cast<const uint4*>(p1 + 16);
            s0 = __builtin_amdgcn_wmma_f32_16x16x32_bf16(
                false, qf[c].v, false, kf0.v, (short)0, s0, false, false);
            s1 = __builtin_amdgcn_wmma_f32_16x16x32_bf16(
                false, qf[c].v, false, kf1.v, (short)0, s1, false, false);
        }

        // ---- online softmax (rows live in elem e + 8*lhalf, cols in lanes) ----
#pragma unroll
        for (int e = 0; e < 8; ++e) {
            const int r = q0 + 8 * lhalf + e;
            float v0 = s0[e] * scale;
            float v1 = s1[e] * scale;
            if (kj + l15 > r)      v0 = -1.0e9f;
            if (kj + 16 + l15 > r) v1 = -1.0e9f;
            float mx = fmaxf(v0, v1);
#pragma unroll
            for (int off = 8; off >= 1; off >>= 1)
                mx = fmaxf(mx, __shfl_xor(mx, off, 16));
            const float mnew = fmaxf(row_m[e], mx);
            const float corr = __expf(row_m[e] - mnew);
            const float p0 = __expf(v0 - mnew);
            const float p1 = __expf(v1 - mnew);
            float rs = p0 + p1;
#pragma unroll
            for (int off = 8; off >= 1; off >>= 1)
                rs += __shfl_xor(rs, off, 16);
            row_l[e] = row_l[e] * corr + rs;
            row_m[e] = mnew;
            o[0][e] *= corr; o[1][e] *= corr; o[2][e] *= corr; o[3][e] *= corr;
            unsigned short* pb = &Pbuf[wave][(8 * lhalf + e) * 40];
            pb[l15]      = f32_to_bf16(p0);
            pb[16 + l15] = f32_to_bf16(p1);
        }
        __builtin_amdgcn_wave_barrier();          // keep LDS store->load ordered

        // ---- P (A layout, 16x32) read back from LDS ----
        FragAB pf;
        {
            const unsigned short* pb = &Pbuf[wave][l15 * 40 + lhalf * 8];
            pf.q[0] = *reinterpret_cast<const uint4*>(pb);
            pf.q[1] = *reinterpret_cast<const uint4*>(pb + 16);
        }
        // ---- O += P * V   (V transposed -> contiguous B fragments) ----
#pragma unroll
        for (int t = 0; t < 4; ++t) {
            FragAB vf;
            const unsigned short* p = Vb + (size_t)(t * 16 + l15) * S_LEN + kj + lhalf * 8;
            vf.q[0] = *reinterpret_cast<const uint4*>(p);
            vf.q[1] = *reinterpret_cast<const uint4*>(p + 16);
            o[t] = __builtin_amdgcn_wmma_f32_16x16x32_bf16(
                false, pf.v, false, vf.v, (short)0, o[t], false, false);
        }
    }

    // ---- normalize + write head-concat bf16 [b, s, h*64+d] ----
    const int b = bh >> 4, h = bh & 15;
#pragma unroll
    for (int t = 0; t < 4; ++t) {
#pragma unroll
        for (int e = 0; e < 8; ++e) {
            const int r = q0 + 8 * lhalf + e;
            const float y = o[t][e] / row_l[e];
            Oh[((size_t)(b * S_LEN + r)) * D_MODEL + h * DK + t * 16 + l15] =
                f32_to_bf16(y);
        }
    }
}

// ---------------------------------------------------------------------------
// Host-side orchestration
// ---------------------------------------------------------------------------
extern "C" void kernel_launch(void* const* d_in, const int* in_sizes, int n_in,
                              void* d_out, int out_size, void* d_ws, size_t ws_size,
                              hipStream_t stream) {
    (void)in_sizes; (void)n_in; (void)out_size; (void)ws_size;
    const float* q  = (const float*)d_in[0];
    const float* k  = (const float*)d_in[1];
    const float* v  = (const float*)d_in[2];
    /* d_in[3] = mask: causal, handled analytically */
    const float* wq = (const float*)d_in[4];
    const float* bq = (const float*)d_in[5];
    const float* wk = (const float*)d_in[6];
    const float* bk = (const float*)d_in[7];
    const float* wv = (const float*)d_in[8];
    const float* bv = (const float*)d_in[9];
    const float* wo = (const float*)d_in[10];
    const float* bo = (const float*)d_in[11];

    const size_t ACT_E = (size_t)BATCH * S_LEN * D_MODEL;   // 4194304
    const size_t W_E   = (size_t)D_MODEL * D_MODEL;         // 1048576

    char* ws = (char*)d_ws;
    unsigned short* qb   = (unsigned short*)(ws);
    unsigned short* kb   = (unsigned short*)(ws + 1 * ACT_E * 2);
    unsigned short* vb   = (unsigned short*)(ws + 2 * ACT_E * 2);
    unsigned short* wqb  = (unsigned short*)(ws + 3 * ACT_E * 2);
    unsigned short* wkb  = (unsigned short*)(ws + 3 * ACT_E * 2 + 1 * W_E * 2);
    unsigned short* wvb  = (unsigned short*)(ws + 3 * ACT_E * 2 + 2 * W_E * 2);
    unsigned short* wob  = (unsigned short*)(ws + 3 * ACT_E * 2 + 3 * W_E * 2);
    unsigned short* Qh   = (unsigned short*)(ws + 3 * ACT_E * 2 + 4 * W_E * 2);
    unsigned short* Khm  = (unsigned short*)(ws + 4 * ACT_E * 2 + 4 * W_E * 2);
    unsigned short* Vth  = (unsigned short*)(ws + 5 * ACT_E * 2 + 4 * W_E * 2);
    unsigned short* attn = (unsigned short*)(ws + 6 * ACT_E * 2 + 4 * W_E * 2);

    // fp32 -> bf16 conversions
    cvt_f32_bf16_kernel<<<2048, 256, 0, stream>>>(q,  qb,  (int)ACT_E);
    cvt_f32_bf16_kernel<<<2048, 256, 0, stream>>>(k,  kb,  (int)ACT_E);
    cvt_f32_bf16_kernel<<<2048, 256, 0, stream>>>(v,  vb,  (int)ACT_E);
    cvt_f32_bf16_kernel<<< 512, 256, 0, stream>>>(wq, wqb, (int)W_E);
    cvt_f32_bf16_kernel<<< 512, 256, 0, stream>>>(wk, wkb, (int)W_E);
    cvt_f32_bf16_kernel<<< 512, 256, 0, stream>>>(wv, wvb, (int)W_E);
    cvt_f32_bf16_kernel<<< 512, 256, 0, stream>>>(wo, wob, (int)W_E);

    // Q/K/V projections (WMMA GEMM + bias, head-friendly layouts)
    dim3 ggrid(D_MODEL / 128, (BATCH * S_LEN) / 128);   // (8, 32)
    gemm_bias_kernel<1><<<ggrid, 256, 0, stream>>>(qb, wqb, bq, (void*)Qh);
    gemm_bias_kernel<1><<<ggrid, 256, 0, stream>>>(kb, wkb, bk, (void*)Khm);
    gemm_bias_kernel<2><<<ggrid, 256, 0, stream>>>(vb, wvb, bv, (void*)Vth);

    // causal flash attention
    flash_attn_kernel<<<dim3(BATCH * NHEAD, S_LEN / 64), 128, 0, stream>>>(
        Qh, Khm, Vth, attn);

    // output projection -> fp32 d_out
    gemm_bias_kernel<0><<<ggrid, 256, 0, stream>>>(attn, wob, bo, d_out);
}
